// VectorQuantizer_3435973836880
// MI455X (gfx1250) — matrix operand order repaired and checked by reading, not compiled, not run on hardware
//
#include <hip/hip_runtime.h>
#include <hip/hip_bf16.h>
#include <math.h>

// ---------------- problem constants ----------------
#define N_E    8192
#define E_DIM  256
#define M_ROWS 16384            // 16 * 32 * 32 flattened pixels
#define LOSS_SCALE 1.25f        // 1 + BETA, BETA = 0.25

typedef __attribute__((ext_vector_type(16))) __bf16 v16bf;
typedef __attribute__((ext_vector_type(8)))  float  v8f;

// ---------------- d_out layout (float elements, reference return order) ----
// loss(1), z_q(16*256*32*32), perplexity(1), min_encodings(16384*8192),
// min_encoding_indices(16384), second_encoding_indices(16384)
#define OFF_LOSS 0
#define OFF_ZQ   1
#define OFF_PERP (1 + 16*256*32*32)                  /* 4194305 */
#define OFF_ENC  (OFF_PERP + 1)                      /* 4194306 */
#define OFF_IDX1 ((size_t)OFF_ENC + (size_t)M_ROWS * (size_t)N_E)
#define OFF_IDX2 (OFF_IDX1 + (size_t)M_ROWS)

// ---------------- workspace layout (byte offsets) ----------------
#define WS_EHI   ((size_t)0)            // 8192*256 bf16  = 4 MB
#define WS_ELO   ((size_t)4194304)      // 8192*256 bf16  = 4 MB
#define WS_ELEN2 ((size_t)8388608)      // 8192 f32
#define WS_IDX1  ((size_t)8421376)      // 16384 i32
#define WS_IDX2  ((size_t)8486912)      // 16384 i32
#define WS_HIST  ((size_t)8552448)      // 8192 i32
#define WS_PART  ((size_t)8585216)      // 4096 f32 partial loss sums

// ===========================================================================
// Kernel 1: embedding -> bf16 hi/lo split, ||e||^2 per row, zero histogram
// grid = N_E blocks x 256 threads (one block per embedding row)
// ===========================================================================
__global__ void __launch_bounds__(256) vq_prep(
    const float* __restrict__ emb, __bf16* __restrict__ ehi,
    __bf16* __restrict__ elo, float* __restrict__ elen2,
    int* __restrict__ hist) {
  const int n = blockIdx.x;
  const int t = threadIdx.x;
  const float v = emb[n * E_DIM + t];
  const __bf16 h = (__bf16)v;
  ehi[n * E_DIM + t] = h;
  elo[n * E_DIM + t] = (__bf16)(v - (float)h);
  __shared__ float red[256];
  red[t] = v * v;
  __syncthreads();
  for (int s = 128; s > 0; s >>= 1) {
    if (t < s) red[t] += red[t + s];
    __syncthreads();
  }
  if (t == 0) { elen2[n] = red[0]; hist[n] = 0; }
}

// ===========================================================================
// Kernel 2: fused bf16x3 WMMA distance GEMM + top-2 argmin
// grid = M_ROWS/16 blocks x 256 threads (8 waves). Block handles 16 rows of
// zf; wave w covers embedding columns [w*1024, w*1024+1024).
// dist'(i,j) = ||e_j||^2 - 2 * dot(zf_i, e_j)   (||z||^2 constant per row)
// ===========================================================================
__global__ void __launch_bounds__(256) vq_search(
    const float* __restrict__ z, const __bf16* __restrict__ ehi,
    const __bf16* __restrict__ elo, const float* __restrict__ elen2,
    int* __restrict__ idx1, int* __restrict__ idx2, int* __restrict__ hist,
    float* __restrict__ out) {
  const int tid  = threadIdx.x;
  const int wave = tid >> 5;
  const int lane = tid & 31;
  const int lm   = lane & 15;     // A row / B column within tile
  const int half = lane >> 4;
  const int m0   = blockIdx.x * 16;

  // ---- A fragments: 16 rows x 256 K of zf, split into bf16 hi/lo, in VGPRs.
  // zf[i][c] = z[b][c][h][w] with i = b*1024 + h*32 + w  (rows of a tile never
  // cross a b boundary since 16 | 1024).
  const int i_row = m0 + lm;
  const int zbase = ((i_row >> 10) << 18) + (i_row & 1023);
  v16bf Ahi[8], Alo[8];
#pragma unroll
  for (int q = 0; q < 8; ++q) {
    float av[16];
#pragma unroll
    for (int e = 0; e < 8; ++e) {
      // A k order used here (contiguous, same as B): lane half 0 -> K = 0..15, half 1 -> K = 16..31
      const int c0 = q * 32 + half * 16 + e;
      const int c1 = q * 32 + half * 16 + 8 + e;
      av[e]     = z[zbase + (c0 << 10)];
      av[8 + e] = z[zbase + (c1 << 10)];
    }
    v16bf hi, lo;
#pragma unroll
    for (int e = 0; e < 16; ++e) {
      const __bf16 h = (__bf16)av[e];
      hi[e] = h;
      lo[e] = (__bf16)(av[e] - (float)h);
    }
    Ahi[q] = hi;
    Alo[q] = lo;
  }

  // ---- running top-2 per C-fragment row (row = half*8 + r, col = nb + lm)
  float bd[8], sd[8];
  int   bi[8], si[8];
#pragma unroll
  for (int r = 0; r < 8; ++r) { bd[r] = 3.0e38f; sd[r] = 3.0e38f; bi[r] = 0; si[r] = 0; }

  const int nwave = wave * 1024;
  for (int it = 0; it < 64; ++it) {
    const int nb = nwave + it * 16;
    const int n  = nb + lm;   // this lane's embedding column (all 8 rows share it)
    v8f acc = {0.f, 0.f, 0.f, 0.f, 0.f, 0.f, 0.f, 0.f};
#pragma unroll
    for (int q = 0; q < 8; ++q) {
      // B k order used here (contiguous, same as A): lane = column n; half 0 -> K 0..15, half 1 -> K 16..31
      // -> 16 contiguous bf16 (32 B) from embedding row n.
      const v16bf bhi = *(const v16bf*)(ehi + (size_t)n * E_DIM + q * 32 + half * 16);
      const v16bf blo = *(const v16bf*)(elo + (size_t)n * E_DIM + q * 32 + half * 16);
      acc = __builtin_amdgcn_wmma_f32_16x16x32_bf16(false, Ahi[q], false, bhi,
                                                    (short)0, acc, false, false);
      acc = __builtin_amdgcn_wmma_f32_16x16x32_bf16(false, Ahi[q], false, blo,
                                                    (short)0, acc, false, false);
      acc = __builtin_amdgcn_wmma_f32_16x16x32_bf16(false, Alo[q], false, bhi,
                                                    (short)0, acc, false, false);
    }
    const float e2 = elen2[n];
#pragma unroll
    for (int r = 0; r < 8; ++r) {
      const float d = __builtin_fmaf(-2.f, acc[r], e2);
      if (d < bd[r])      { sd[r] = bd[r]; si[r] = bi[r]; bd[r] = d; bi[r] = n; }
      else if (d < sd[r]) { sd[r] = d; si[r] = n; }
    }
  }

  // ---- merge top-2 across the 16 lanes per row and across the 8 waves ----
  __shared__ float sBD[2048], sSD[2048];
  __shared__ int   sBI[2048], sSI[2048];
  const int base = tid * 8;
#pragma unroll
  for (int r = 0; r < 8; ++r) {
    sBD[base + r] = bd[r]; sBI[base + r] = bi[r];
    sSD[base + r] = sd[r]; sSI[base + r] = si[r];
  }
  __syncthreads();
  if (tid < 16) {
    const int m = tid;                 // row within tile
    const int lh = (m >> 3) << 4;      // lane half holding this row
    const int r  = m & 7;
    float BD = 3.0e38f, SD = 3.0e38f;
    int BI = 0, SI = 0;
    for (int w = 0; w < 8; ++w) {      // wave order = ascending n chunks
      for (int j = 0; j < 16; ++j) {
        const int cb = ((w * 32) + lh + j) * 8 + r;
        const float d0 = sBD[cb]; const int i0 = sBI[cb];
        if (d0 < BD)      { SD = BD; SI = BI; BD = d0; BI = i0; }
        else if (d0 < SD) { SD = d0; SI = i0; }
        const float d1 = sSD[cb]; const int i1 = sSI[cb];
        if (d1 < BD)      { SD = BD; SI = BI; BD = d1; BI = i1; }
        else if (d1 < SD) { SD = d1; SI = i1; }
      }
    }
    const int gi = m0 + m;
    idx1[gi] = BI;
    idx2[gi] = SI;
    out[OFF_IDX1 + gi] = (float)BI;
    out[OFF_IDX2 + gi] = (float)SI;
    atomicAdd(&hist[BI], 1);           // integer atomics: order-independent
  }
}

// ===========================================================================
// Kernel 3: min_encodings one-hot fill (536 MB stream, the HBM-bound part)
// grid = M_ROWS blocks x 256 threads; 8-byte stores (ENC offset is 8B-aligned)
// ===========================================================================
__global__ void __launch_bounds__(256) vq_fill_enc(
    const int* __restrict__ idx1, float* __restrict__ out) {
  const int i = blockIdx.x;
  const int t = threadIdx.x;
  const int idx = idx1[i];
  float2* row = (float2*)(out + OFF_ENC + (size_t)i * N_E);
#pragma unroll
  for (int rep = 0; rep < 16; ++rep) {
    const int j2 = rep * 256 + t;
    const int cb = j2 << 1;
    float2 v;
    v.x = (cb == idx)     ? 1.f : 0.f;
    v.y = (cb + 1 == idx) ? 1.f : 0.f;
    row[j2] = v;
  }
}

// ===========================================================================
// Kernel 4: z_q gather (coalesced writes in NCHW) + partial loss sums
// grid = 16*256 blocks (one per (b, c) plane) x 256 threads
// ===========================================================================
__global__ void __launch_bounds__(256) vq_zq_loss(
    const float* __restrict__ z, const float* __restrict__ emb,
    const int* __restrict__ idx1, float* __restrict__ out,
    float* __restrict__ part) {
  const int bc = blockIdx.x;
  const int b = bc >> 8, c = bc & 255;
  const int t = threadIdx.x;
  const int base = (b << 18) + (c << 10);
  float s = 0.f;
#pragma unroll
  for (int rep = 0; rep < 4; ++rep) {
    const int pos = rep * 256 + t;
    const int i = (b << 10) + pos;
    const int idx = idx1[i];
    const float zq = emb[idx * E_DIM + c];     // gather, L2-resident
    const float zv = z[base + pos];
    out[OFF_ZQ + base + pos] = zq;             // straight-through: forward = z_q
    const float df = zq - zv;
    s += df * df;
  }
  __shared__ float red[256];
  red[t] = s;
  __syncthreads();
  for (int k = 128; k > 0; k >>= 1) {
    if (t < k) red[t] += red[t + k];
    __syncthreads();
  }
  if (t == 0) part[bc] = red[0];
}

// ===========================================================================
// Kernel 5: loss + perplexity finalization (single block, deterministic)
// ===========================================================================
__global__ void __launch_bounds__(256) vq_final(
    const float* __restrict__ part, const int* __restrict__ hist,
    float* __restrict__ out) {
  const int t = threadIdx.x;
  __shared__ float red[256];
  float s = 0.f;
  for (int rep = 0; rep < 16; ++rep) s += part[rep * 256 + t];
  red[t] = s;
  __syncthreads();
  for (int k = 128; k > 0; k >>= 1) {
    if (t < k) red[t] += red[t + k];
    __syncthreads();
  }
  if (t == 0) out[OFF_LOSS] = LOSS_SCALE * red[0] / (float)(M_ROWS * E_DIM);
  __syncthreads();
  float sp = 0.f;
  for (int rep = 0; rep < 32; ++rep) {
    const int j = rep * 256 + t;
    const float p = (float)hist[j] * (1.f / (float)M_ROWS);
    sp += p * logf(p + 1e-10f);
  }
  red[t] = sp;
  __syncthreads();
  for (int k = 128; k > 0; k >>= 1) {
    if (t < k) red[t] += red[t + k];
    __syncthreads();
  }
  if (t == 0) out[OFF_PERP] = expf(-red[0]);
}

// ===========================================================================
extern "C" void kernel_launch(void* const* d_in, const int* in_sizes, int n_in,
                              void* d_out, int out_size, void* d_ws, size_t ws_size,
                              hipStream_t stream) {
  const float* z   = (const float*)d_in[0];   // (16, 256, 32, 32) f32
  const float* emb = (const float*)d_in[1];   // (8192, 256) f32
  float* out = (float*)d_out;
  char*  ws  = (char*)d_ws;

  __bf16* ehi   = (__bf16*)(ws + WS_EHI);
  __bf16* elo   = (__bf16*)(ws + WS_ELO);
  float*  elen2 = (float*)(ws + WS_ELEN2);
  int*    idx1  = (int*)(ws + WS_IDX1);
  int*    idx2  = (int*)(ws + WS_IDX2);
  int*    hist  = (int*)(ws + WS_HIST);
  float*  part  = (float*)(ws + WS_PART);

  vq_prep<<<N_E, 256, 0, stream>>>(emb, ehi, elo, elen2, hist);
  vq_search<<<M_ROWS / 16, 256, 0, stream>>>(z, ehi, elo, elen2, idx1, idx2, hist, out);
  vq_fill_enc<<<M_ROWS, 256, 0, stream>>>(idx1, out);
  vq_zq_loss<<<16 * 256, 256, 0, stream>>>(z, emb, idx1, out, part);
  vq_final<<<1, 256, 0, stream>>>(part, hist, out);
}
